// ProteinMPNN_32573031973441
// MI455X (gfx1250) — matrix operand order repaired and checked by reading, not compiled, not run on hardware
//
// ProteinMPNN forward for MI455X (gfx1250), wave32 + WMMA f16->f32.
// All dense layers (edge_emb, W_e, message MLPs, FFNs) run as
// v_wmma_f32_16x16x32_f16 tile loops: 8 waves/block, each wave owns a
// 16x16 output tile.  A tiles staged in LDS (gather/concat fused; two
// contiguous v8h runs per fragment -> ds_load_b128 pairs).  Weights are
// pre-packed into WMMA B-fragment order so each wave fetches its whole
// 16-half B operand as one 32B vector load (2x global_load_b128).
// Workspace use (~151MB): atoms, topk, hE16 (64MB), msg16 (64MB), hV buffers,
// packed f16 weight pool (~3.3MB, L2-resident).

#include <hip/hip_runtime.h>
#include <math.h>

typedef __attribute__((ext_vector_type(16))) _Float16 v16h;
typedef __attribute__((ext_vector_type(8)))  _Float16 v8h;
typedef __attribute__((ext_vector_type(8)))  float    v8f;

#define HD 128
#define KN 32

// ---------------- device helpers ----------------

__device__ __forceinline__ float gelu_f(float x) {
  return 0.5f * x * (1.0f + erff(x * 0.70710678118654752f));
}

// ISA VGPR layouts (CDNA5 ISA):
//   A 16x32 f16: lane m=lane&15, grp=lane>>4; elem e -> K = (e&8)*2 + grp*8 + (e&7)
//     -> two contiguous 8-half runs at K = grp*8 and K = 16 + grp*8.
//   B 32x16 f16: lane n=lane&15, grp=lane>>4; elem e -> K = (e&8)*2 + grp*8 + (e&7) (same as A)
//     -> pre-packed tile-major: [(kt*NT + nt)*32 + lane][16 halfs].
//   C/D 16x16 f32: elem r -> row r + 8*(lane>>4), col lane&15.
__device__ __forceinline__ v8f wmma_tile(const _Float16* __restrict__ As, int lda,
                                         const _Float16* __restrict__ Wp, int NT,
                                         int Kdim, int nt, int lane) {
  v8f acc = {};
  const int mrow = lane & 15;
  const int grp  = lane >> 4;
  const _Float16* arow = As + mrow * lda + grp * 8;
  const _Float16* wl   = Wp + ((size_t)nt * 32 + lane) * 16;
  const size_t kstride = (size_t)NT * 32 * 16;
  int kt = 0;
  for (int kk = 0; kk < Kdim; kk += 32, ++kt) {
    v8h lo = *(const v8h*)(arow + kk);
    v8h hi = *(const v8h*)(arow + kk + 16);
    v16h a = __builtin_shufflevector(lo, hi, 0, 1, 2, 3, 4, 5, 6, 7,
                                             8, 9, 10, 11, 12, 13, 14, 15);
    v16h b = *(const v16h*)(wl + (size_t)kt * kstride);
    if (kk + 32 < Kdim)
      __builtin_prefetch(wl + (size_t)(kt + 1) * kstride, 0, 1);
    acc = __builtin_amdgcn_wmma_f32_16x16x32_f16(false, a, false, b, (short)0, acc,
                                                 false, false);
  }
  return acc;
}

__device__ __forceinline__ void store_tile_f16(_Float16* dst, int ldd, v8f acc,
                                               const float* bias, int n0, int lane,
                                               bool act) {
  int n = lane & 15, grp = lane >> 4;
  float bv = bias ? bias[n0 + n] : 0.0f;
#pragma unroll
  for (int r = 0; r < 8; ++r) {
    float x = acc[r] + bv;
    if (act) x = gelu_f(x);
    dst[(r + 8 * grp) * ldd + n0 + n] = (_Float16)x;
  }
}

__device__ __forceinline__ void store_tile_f32(float* dst, int ldd, v8f acc,
                                               const float* bias, int n0, int lane) {
  int n = lane & 15, grp = lane >> 4;
  float bv = bias ? bias[n0 + n] : 0.0f;
#pragma unroll
  for (int r = 0; r < 8; ++r) dst[(r + 8 * grp) * ldd + n0 + n] = acc[r] + bv;
}

__device__ __forceinline__ void row_layernorm(float* row, const float* g,
                                              const float* b) {
  float m = 0.f;
  for (int c = 0; c < HD; ++c) m += row[c];
  m *= (1.0f / HD);
  float v = 0.f;
  for (int c = 0; c < HD; ++c) { float d = row[c] - m; v += d * d; }
  v *= (1.0f / HD);
  float inv = rsqrtf(v + 1e-5f);
  for (int c = 0; c < HD; ++c) row[c] = (row[c] - m) * inv * g[c] + b[c];
}

// ---------------- small utility kernels ----------------

// f32 [din][dout] -> f16 packed in WMMA B-fragment order:
//   dst[(((kt*NT + nt)*32) + grp*16 + n)*16 + e],  row = kt*32 + (e&8)*2 + grp*8 + (e&7),
//   col = nt*16 + n,  NT = dout/16.
__global__ void k_cvt_pack(const float* __restrict__ s, _Float16* __restrict__ d,
                           int din, int dout) {
  int idx = blockIdx.x * blockDim.x + threadIdx.x;
  if (idx >= din * dout) return;
  int row = idx / dout, col = idx - row * dout;
  int kt = row >> 5, kb = row & 31;
  int grp = (kb >> 3) & 1, e = (kb & 7) | ((kb >> 4) << 3);
  int nt = col >> 4, n = col & 15;
  int NT = dout >> 4;
  d[((((size_t)kt * NT + nt) * 32) + grp * 16 + n) * 16 + e] = (_Float16)s[idx];
}

// atoms: N, Ca, C, O (from X) + virtual Cb; layout atoms[(atom*BL + node)*3 + d]
__global__ void k_atoms(const float* __restrict__ X, float* __restrict__ atoms,
                        int BL) {
  int i = blockIdx.x * blockDim.x + threadIdx.x;
  if (i >= BL) return;
  const float* x = X + (size_t)i * 12;
  float N[3], CA[3], CC[3], OO[3], bv[3], cv[3];
  for (int d = 0; d < 3; ++d) {
    N[d] = x[d]; CA[d] = x[3 + d]; CC[d] = x[6 + d]; OO[d] = x[9 + d];
  }
  for (int d = 0; d < 3; ++d) { bv[d] = CA[d] - N[d]; cv[d] = CC[d] - CA[d]; }
  float av[3] = { bv[1]*cv[2] - bv[2]*cv[1],
                  bv[2]*cv[0] - bv[0]*cv[2],
                  bv[0]*cv[1] - bv[1]*cv[0] };
  for (int d = 0; d < 3; ++d) {
    float cb = -0.58273431f*av[d] + 0.56802827f*bv[d] - 0.54067466f*cv[d] + CA[d];
    atoms[((size_t)0*BL + i)*3 + d] = N[d];
    atoms[((size_t)1*BL + i)*3 + d] = CA[d];
    atoms[((size_t)2*BL + i)*3 + d] = CC[d];
    atoms[((size_t)3*BL + i)*3 + d] = OO[d];
    atoms[((size_t)4*BL + i)*3 + d] = cb;
  }
}

// top-32 smallest masked Ca-Ca distances per node (matches jax top_k tie-break)
__global__ __launch_bounds__(256) void k_topk(const float* __restrict__ atoms,
                                              const float* __restrict__ mask,
                                              float* __restrict__ Dnb,
                                              int* __restrict__ Eidx, int L, int BL) {
  __shared__ float dist[1024];
  __shared__ float redv[256];
  __shared__ int   redi[256];
  __shared__ float sDmax;
  int row = blockIdx.x, tid = threadIdx.x;
  int b = row / L;
  const float* Ca = atoms + (size_t)BL * 3;  // atom 1
  float cx = Ca[(size_t)row*3+0], cy = Ca[(size_t)row*3+1], cz = Ca[(size_t)row*3+2];
  float mi = mask[row];
  float lmax = -1e30f;
  for (int j = tid; j < L; j += 256) {
    int gj = b * L + j;
    float dx = cx - Ca[(size_t)gj*3+0];
    float dy = cy - Ca[(size_t)gj*3+1];
    float dz = cz - Ca[(size_t)gj*3+2];
    float D = mi * mask[gj] * sqrtf(dx*dx + dy*dy + dz*dz + 1e-6f);
    dist[j] = D;
    lmax = fmaxf(lmax, D);
  }
  redv[tid] = lmax; __syncthreads();
  for (int st = 128; st > 0; st >>= 1) {
    if (tid < st) redv[tid] = fmaxf(redv[tid], redv[tid + st]);
    __syncthreads();
  }
  if (tid == 0) sDmax = redv[0];
  __syncthreads();
  for (int j = tid; j < L; j += 256) {
    int gj = b * L + j;
    dist[j] += (1.0f - mi * mask[gj]) * sDmax;
  }
  __syncthreads();
  for (int r = 0; r < KN; ++r) {
    float best = 1e30f; int bi = 0x7fffffff;
    for (int j = tid; j < L; j += 256) {
      float v = dist[j];
      if (v < best || (v == best && j < bi)) { best = v; bi = j; }
    }
    redv[tid] = best; redi[tid] = bi; __syncthreads();
    for (int st = 128; st > 0; st >>= 1) {
      if (tid < st) {
        float v2 = redv[tid + st]; int i2 = redi[tid + st];
        if (v2 < redv[tid] || (v2 == redv[tid] && i2 < redi[tid])) {
          redv[tid] = v2; redi[tid] = i2;
        }
      }
      __syncthreads();
    }
    if (tid == 0) {
      Dnb[(size_t)row * KN + r] = redv[0];
      Eidx[(size_t)row * KN + r] = redi[0];
      dist[redi[0]] = 1e30f;
    }
    __syncthreads();
  }
}

__global__ void k_hs(const int* __restrict__ S, const float* __restrict__ Ws,
                     _Float16* __restrict__ hS16, int BL) {
  int t = blockIdx.x * blockDim.x + threadIdx.x;
  if (t >= BL * HD) return;
  int node = t >> 7, c = t & 127;
  hS16[t] = (_Float16)Ws[S[node] * HD + c];
}

__global__ __launch_bounds__(256) void k_rank(const float* __restrict__ chainM,
                                              const float* __restrict__ mask,
                                              const float* __restrict__ randn,
                                              int* __restrict__ rank, int L) {
  __shared__ int red[256];
  int row = blockIdx.x, tid = threadIdx.x;
  int b = row / L, i = row % L;
  float ki = (chainM[row] * mask[row] + 1e-4f) * fabsf(randn[row]);
  int cnt = 0;
  for (int j = tid; j < L; j += 256) {
    int gj = b * L + j;
    float kj = (chainM[gj] * mask[gj] + 1e-4f) * fabsf(randn[gj]);
    if (kj < ki || (kj == ki && j < i)) cnt++;
  }
  red[tid] = cnt; __syncthreads();
  for (int st = 128; st > 0; st >>= 1) {
    if (tid < st) red[tid] += red[tid + st];
    __syncthreads();
  }
  if (tid == 0) rank[row] = red[0];
}

__global__ void k_mar(const int* __restrict__ rank, const int* __restrict__ Eidx,
                      float* __restrict__ mar, int L, int NE) {
  int e = blockIdx.x * blockDim.x + threadIdx.x;
  if (e >= NE) return;
  int i = e >> 5;
  int b = i / L;
  int j = b * L + Eidx[e];
  mar[e] = (rank[i] > rank[j]) ? 1.0f : 0.0f;
}

// ---------------- feature + edge embedding (fused: feat -> edge_emb -> LN -> W_e) --

__constant__ int c_pa[25] = {1, 0,2,3,4, 1,1,1,1, 0,0,0, 4,4,3, 0,2,3,4, 2,3,4, 2,3,2};
__constant__ int c_pb[25] = {1, 0,2,3,4, 0,2,3,4, 2,3,4, 2,3,2, 1,1,1,1, 0,0,0, 4,4,3};

__global__ __launch_bounds__(256) void k_edge_features(
    const float* __restrict__ atoms, const float* __restrict__ Dnb,
    const int* __restrict__ Eidx, const int* __restrict__ residx,
    const int* __restrict__ chains,
    const float* __restrict__ pe_w, const float* __restrict__ pe_b,
    const _Float16* __restrict__ Wedge, const float* __restrict__ ne_g,
    const float* __restrict__ ne_b,
    const _Float16* __restrict__ We, const float* __restrict__ We_b,
    _Float16* __restrict__ hE16, int L, int BL) {
  __shared__ _Float16 sA[16 * 416];
  __shared__ float sC[16 * 128];
  __shared__ float sD[16 * 25];
  __shared__ int sEnc[16];
  int tid = threadIdx.x, lane = tid & 31, wv = tid >> 5;
  int e0 = blockIdx.x * 16;
  for (int t = tid; t < 16 * 25; t += 256) {
    int r = t / 25, p = t - r * 25;
    int e = e0 + r;
    int i = e >> 5;
    int b = i / L;
    int j = b * L + Eidx[e];
    float D;
    if (p == 0) {
      D = Dnb[e];
    } else {
      const float* A  = atoms + ((size_t)c_pa[p] * BL + i) * 3;
      const float* Bt = atoms + ((size_t)c_pb[p] * BL + j) * 3;
      float dx = A[0]-Bt[0], dy = A[1]-Bt[1], dz = A[2]-Bt[2];
      D = sqrtf(dx*dx + dy*dy + dz*dz + 1e-6f);
    }
    sD[t] = D;
  }
  if (tid < 16) {
    int e = e0 + tid;
    int i = e >> 5;
    int b = i / L;
    int j = b * L + Eidx[e];
    int off = residx[i] - residx[j];
    int ec = (chains[i] == chains[j]) ? 1 : 0;
    int v = off + 32;
    v = v < 0 ? 0 : (v > 64 ? 64 : v);
    sEnc[tid] = ec ? v : 65;
  }
  __syncthreads();
  for (int t = tid; t < 16 * 416; t += 256) {
    int r = t / 416, f = t - r * 416;
    float val;
    if (f < 16) {
      val = pe_w[sEnc[r] * 16 + f] + pe_b[f];
    } else {
      int p = (f - 16) >> 4, m = (f - 16) & 15;
      float mu = 2.0f + (20.0f / 15.0f) * (float)m;
      float x = (sD[r * 25 + p] - mu) * (1.0f / 1.25f);
      val = expf(-x * x);
    }
    sA[t] = (_Float16)val;
  }
  __syncthreads();
  v8f acc = wmma_tile(sA, 416, Wedge, 8, 416, wv, lane);   // edge_emb (no bias)
  int n0 = wv * 16;
  store_tile_f32(sC, HD, acc, nullptr, n0, lane);
  __syncthreads();
  if (tid < 16) row_layernorm(sC + tid * HD, ne_g, ne_b);  // norm_edges
  __syncthreads();
  for (int t = tid; t < 16 * HD; t += 256) sA[t] = (_Float16)sC[t];
  __syncthreads();
  acc = wmma_tile(sA, HD, We, 8, HD, wv, lane);            // W_e
  int n = lane & 15, grp = lane >> 4;
  float bb = We_b[n0 + n];
#pragma unroll
  for (int r = 0; r < 8; ++r)
    hE16[(size_t)(e0 + r + 8 * grp) * HD + n0 + n] = (_Float16)(acc[r] + bb);
}

// ---------------- encoder message: [hV_i, hE, hV_j] -> 3-layer MLP, masked ----

__global__ __launch_bounds__(256) void k_enc_msg(
    const _Float16* __restrict__ hV16, const _Float16* __restrict__ hE16,
    const int* __restrict__ Eidx, const float* __restrict__ mask,
    const _Float16* __restrict__ W1, const float* __restrict__ b1,
    const _Float16* __restrict__ W2, const float* __restrict__ b2,
    const _Float16* __restrict__ W3, const float* __restrict__ b3,
    _Float16* __restrict__ msg, int L) {
  __shared__ _Float16 sA[16 * 384];
  __shared__ _Float16 sB[16 * 128];
  __shared__ int sI[16], sJ[16];
  __shared__ float sMA[16];
  int tid = threadIdx.x, lane = tid & 31, wv = tid >> 5;
  int e0 = blockIdx.x * 16;
  if (tid < 16) {
    int e = e0 + tid;
    int i = e >> 5;
    int b = i / L;
    int j = b * L + Eidx[e];
    sI[tid] = i; sJ[tid] = j;
    sMA[tid] = mask[i] * mask[j];
  }
  __syncthreads();
  for (int t = tid; t < 16 * 384; t += 256) {
    int r = t / 384, c = t - r * 384;
    _Float16 v;
    if (c < 128)      v = hV16[(size_t)sI[r] * HD + c];
    else if (c < 256) v = hE16[(size_t)(e0 + r) * HD + (c - 128)];
    else              v = hV16[(size_t)sJ[r] * HD + (c - 256)];
    sA[t] = v;
  }
  __syncthreads();
  int n0 = wv * 16;
  v8f acc = wmma_tile(sA, 384, W1, 8, 384, wv, lane);
  store_tile_f16(sB, HD, acc, b1, n0, lane, true);
  __syncthreads();
  acc = wmma_tile(sB, HD, W2, 8, HD, wv, lane);
  store_tile_f16(sA, HD, acc, b2, n0, lane, true);   // reuse sA front, lda=HD
  __syncthreads();
  acc = wmma_tile(sA, HD, W3, 8, HD, wv, lane);
  int n = lane & 15, grp = lane >> 4;
  float bb = b3[n0 + n];
#pragma unroll
  for (int r = 0; r < 8; ++r) {
    int m = r + 8 * grp;
    msg[(size_t)(e0 + m) * HD + n0 + n] = (_Float16)((acc[r] + bb) * sMA[m]);
  }
}

// ---------------- node update: sum_k msg /30 + residual + LN(norm1) ----------

__global__ __launch_bounds__(128) void k_node_update(
    const _Float16* __restrict__ msg, const float* __restrict__ hVin,
    const float* __restrict__ gam, const float* __restrict__ bet,
    float* __restrict__ out32, _Float16* __restrict__ out16) {
  __shared__ float red[128];
  int node = blockIdx.x, c = threadIdx.x;
  float s = 0.f;
  const _Float16* mp = msg + ((size_t)node * KN) * HD + c;
  for (int k = 0; k < KN; ++k) s += (float)mp[k * HD];
  float u = hVin[(size_t)node * HD + c] + s * (1.0f / 30.0f);
  red[c] = u; __syncthreads();
  for (int st = 64; st > 0; st >>= 1) { if (c < st) red[c] += red[c + st]; __syncthreads(); }
  float mean = red[0] * (1.0f / HD);
  __syncthreads();
  float d = u - mean;
  red[c] = d * d; __syncthreads();
  for (int st = 64; st > 0; st >>= 1) { if (c < st) red[c] += red[c + st]; __syncthreads(); }
  float var = red[0] * (1.0f / HD);
  float y = d * rsqrtf(var + 1e-5f) * gam[c] + bet[c];
  out32[(size_t)node * HD + c] = y;
  out16[(size_t)node * HD + c] = (_Float16)y;
}

// ---------------- position-wise FFN + LN(norm2) + mask ----------------------

__global__ __launch_bounds__(256) void k_ffn(
    const _Float16* __restrict__ in16, const float* __restrict__ in32,
    const _Float16* __restrict__ Wi, const float* __restrict__ bi,
    const _Float16* __restrict__ Wo, const float* __restrict__ bo,
    const float* __restrict__ n2g, const float* __restrict__ n2b,
    const float* __restrict__ mask,
    float* __restrict__ out32, _Float16* __restrict__ out16) {
  __shared__ _Float16 sA[16 * 128];
  __shared__ _Float16 sB[16 * 512];
  __shared__ float sC[16 * 128];
  int tid = threadIdx.x, lane = tid & 31, wv = tid >> 5;
  int v0 = blockIdx.x * 16;
  for (int t = tid; t < 16 * HD; t += 256) sA[t] = in16[(size_t)v0 * HD + t];
  __syncthreads();
#pragma unroll
  for (int q = 0; q < 4; ++q) {
    int ntq = wv + 8 * q;
    v8f accq = wmma_tile(sA, HD, Wi, 32, HD, ntq, lane);
    store_tile_f16(sB, 512, accq, bi, ntq * 16, lane, true);
  }
  __syncthreads();
  int n0 = wv * 16;
  v8f acc = wmma_tile(sB, 512, Wo, 8, 512, wv, lane);
  store_tile_f32(sC, HD, acc, bo, n0, lane);
  __syncthreads();
  if (tid < 16) {
    float* row = sC + tid * HD;
    const float* res = in32 + (size_t)(v0 + tid) * HD;
    for (int c = 0; c < HD; ++c) row[c] += res[c];
    row_layernorm(row, n2g, n2b);
    float mk = mask[v0 + tid];
    for (int c = 0; c < HD; ++c) row[c] *= mk;
  }
  __syncthreads();
  for (int t = tid; t < 16 * HD; t += 256) {
    float y = sC[t];
    out32[(size_t)v0 * HD + t] = y;
    out16[(size_t)v0 * HD + t] = (_Float16)y;
  }
}

// ---------------- encoder edge update: residual + LN(norm3), in-place hE16 ---

__global__ __launch_bounds__(256) void k_enc_edge(
    const _Float16* __restrict__ hV16, _Float16* __restrict__ hE16,
    const int* __restrict__ Eidx,
    const _Float16* __restrict__ W1, const float* __restrict__ b1,
    const _Float16* __restrict__ W2, const float* __restrict__ b2,
    const _Float16* __restrict__ W3, const float* __restrict__ b3,
    const float* __restrict__ n3g, const float* __restrict__ n3b, int L) {
  __shared__ _Float16 sA[16 * 384];
  __shared__ _Float16 sB[16 * 128];
  __shared__ float sC[16 * 128];
  __shared__ int sI[16], sJ[16];
  int tid = threadIdx.x, lane = tid & 31, wv = tid >> 5;
  int e0 = blockIdx.x * 16;
  if (tid < 16) {
    int e = e0 + tid;
    int i = e >> 5;
    int b = i / L;
    sI[tid] = i; sJ[tid] = b * L + Eidx[e];
  }
  __syncthreads();
  for (int t = tid; t < 16 * 384; t += 256) {
    int r = t / 384, c = t - r * 384;
    _Float16 v;
    if (c < 128)      v = hV16[(size_t)sI[r] * HD + c];
    else if (c < 256) v = hE16[(size_t)(e0 + r) * HD + (c - 128)];
    else              v = hV16[(size_t)sJ[r] * HD + (c - 256)];
    sA[t] = v;
  }
  __syncthreads();
  int n0 = wv * 16;
  v8f acc = wmma_tile(sA, 384, W1, 8, 384, wv, lane);
  store_tile_f16(sB, HD, acc, b1, n0, lane, true);
  __syncthreads();
  acc = wmma_tile(sB, HD, W2, 8, HD, wv, lane);
  store_tile_f16(sA, HD, acc, b2, n0, lane, true);
  __syncthreads();
  acc = wmma_tile(sA, HD, W3, 8, HD, wv, lane);
  store_tile_f32(sC, HD, acc, b3, n0, lane);
  __syncthreads();
  if (tid < 16) {
    float* row = sC + tid * HD;
    const _Float16* ep = hE16 + (size_t)(e0 + tid) * HD;
    for (int c = 0; c < HD; ++c) row[c] += (float)ep[c];
    row_layernorm(row, n3g, n3b);
  }
  __syncthreads();
  for (int t = tid; t < 16 * HD; t += 256)
    hE16[(size_t)e0 * HD + t] = (_Float16)sC[t];
}

// ---------------- decoder message: [hV_i, m*hE, bw*hS_j, bw*hV_j+fw*hVenc_j] -

__global__ __launch_bounds__(256) void k_dec_msg(
    const _Float16* __restrict__ hV16, const _Float16* __restrict__ hVenc16,
    const _Float16* __restrict__ hE16, const _Float16* __restrict__ hS16,
    const int* __restrict__ Eidx, const float* __restrict__ mask,
    const float* __restrict__ mar,
    const _Float16* __restrict__ W1, const float* __restrict__ b1,
    const _Float16* __restrict__ W2, const float* __restrict__ b2,
    const _Float16* __restrict__ W3, const float* __restrict__ b3,
    _Float16* __restrict__ msg, int L) {
  __shared__ _Float16 sA[16 * 512];
  __shared__ _Float16 sB[16 * 128];
  __shared__ int sI[16], sJ[16];
  __shared__ float sMi[16], sBw[16], sFw[16];
  int tid = threadIdx.x, lane = tid & 31, wv = tid >> 5;
  int e0 = blockIdx.x * 16;
  if (tid < 16) {
    int e = e0 + tid;
    int i = e >> 5;
    int b = i / L;
    int j = b * L + Eidx[e];
    float mi = mask[i], ar = mar[e];
    sI[tid] = i; sJ[tid] = j;
    sMi[tid] = mi; sBw[tid] = mi * ar; sFw[tid] = mi * (1.0f - ar);
  }
  __syncthreads();
  for (int t = tid; t < 16 * 512; t += 256) {
    int r = t >> 9, c = t & 511;
    float v;
    if (c < 128)      v = (float)hV16[(size_t)sI[r] * HD + c];
    else if (c < 256) v = sMi[r] * (float)hE16[(size_t)(e0 + r) * HD + (c - 128)];
    else if (c < 384) v = sBw[r] * (float)hS16[(size_t)sJ[r] * HD + (c - 256)];
    else              v = sBw[r] * (float)hV16[(size_t)sJ[r] * HD + (c - 384)]
                        + sFw[r] * (float)hVenc16[(size_t)sJ[r] * HD + (c - 384)];
    sA[t] = (_Float16)v;
  }
  __syncthreads();
  int n0 = wv * 16;
  v8f acc = wmma_tile(sA, 512, W1, 8, 512, wv, lane);
  store_tile_f16(sB, HD, acc, b1, n0, lane, true);
  __syncthreads();
  acc = wmma_tile(sB, HD, W2, 8, HD, wv, lane);
  store_tile_f16(sA, HD, acc, b2, n0, lane, true);
  __syncthreads();
  acc = wmma_tile(sA, HD, W3, 8, HD, wv, lane);
  int n = lane & 15, grp = lane >> 4;
  float bb = b3[n0 + n];
#pragma unroll
  for (int r = 0; r < 8; ++r)
    msg[(size_t)(e0 + r + 8 * grp) * HD + n0 + n] = (_Float16)(acc[r] + bb);
}

// ---------------- output head: W_out + log_softmax(21) ----------------------

__global__ __launch_bounds__(32) void k_out(const float* __restrict__ hV,
                                            const float* __restrict__ Ww,
                                            const float* __restrict__ Wb,
                                            float* __restrict__ out) {
  __shared__ float lg[21];
  __shared__ float sLse;
  int node = blockIdx.x, t = threadIdx.x;
  if (t < 21) {
    float s = Wb[t];
    const float* h = hV + (size_t)node * HD;
    for (int c = 0; c < HD; ++c) s += h[c] * Ww[c * 21 + t];
    lg[t] = s;
  }
  __syncthreads();
  if (t == 0) {
    float mx = lg[0];
    for (int q = 1; q < 21; ++q) mx = fmaxf(mx, lg[q]);
    float se = 0.f;
    for (int q = 0; q < 21; ++q) se += expf(lg[q] - mx);
    sLse = mx + logf(se);
  }
  __syncthreads();
  if (t < 21) out[(size_t)node * 21 + t] = lg[t] - sLse;
}

// ---------------- host orchestration ----------------------------------------

extern "C" void kernel_launch(void* const* d_in, const int* in_sizes, int n_in,
                              void* d_out, int out_size, void* d_ws, size_t ws_size,
                              hipStream_t stream) {
  (void)n_in; (void)out_size; (void)ws_size;
  const float* X      = (const float*)d_in[0];
  const int*   S      = (const int*)d_in[1];
  const float* mask   = (const float*)d_in[2];
  const float* chainM = (const float*)d_in[3];
  const int*   residx = (const int*)d_in[4];
  const int*   chains = (const int*)d_in[5];
  const float* randn  = (const float*)d_in[6];

  const int L  = 1024;
  const int BL = in_sizes[1];      // B*L
  const int NE = BL * KN;

  // params leaves in jax tree_flatten order (dict keys sorted).
  int pidx = 7;
  auto nx = [&]() { return (const float*)d_in[pidx++]; };
  const float* We_b   = nx(); const float* We_w   = nx();
  const float* Wout_b = nx(); const float* Wout_w = nx();
  const float* Ws     = nx();
  struct DecP { const float *fib,*fiw,*fob,*fow,*n1b,*n1g,*n2b,*n2g,*w1b,*w1w,*w2b,*w2w,*w3b,*w3w; } dp[3];
  for (int l = 0; l < 3; ++l) {
    dp[l].fib = nx(); dp[l].fiw = nx(); dp[l].fob = nx(); dp[l].fow = nx();
    dp[l].n1b = nx(); dp[l].n1g = nx(); dp[l].n2b = nx(); dp[l].n2g = nx();
    dp[l].w1b = nx(); dp[l].w1w = nx(); dp[l].w2b = nx(); dp[l].w2w = nx();
    dp[l].w3b = nx(); dp[l].w3w = nx();
  }
  const float* edge_w = nx();
  struct EncP { const float *fib,*fiw,*fob,*fow,*n1b,*n1g,*n2b,*n2g,*n3b,*n3g,
                *w1b,*w1w,*w11b,*w11w,*w12b,*w12w,*w13b,*w13w,*w2b,*w2w,*w3b,*w3w; } ep[3];
  for (int l = 0; l < 3; ++l) {
    ep[l].fib = nx(); ep[l].fiw = nx(); ep[l].fob = nx(); ep[l].fow = nx();
    ep[l].n1b = nx(); ep[l].n1g = nx(); ep[l].n2b = nx(); ep[l].n2g = nx();
    ep[l].n3b = nx(); ep[l].n3g = nx();
    ep[l].w1b = nx(); ep[l].w1w = nx();
    ep[l].w11b = nx(); ep[l].w11w = nx();
    ep[l].w12b = nx(); ep[l].w12w = nx();
    ep[l].w13b = nx(); ep[l].w13w = nx();
    ep[l].w2b = nx(); ep[l].w2w = nx();
    ep[l].w3b = nx(); ep[l].w3w = nx();
  }
  const float* ne_b = nx(); const float* ne_g = nx();
  const float* pe_b = nx(); const float* pe_w = nx();

  // workspace carve-out
  size_t off = 0;
  auto alloc = [&](size_t bytes) -> void* {
    off = (off + 255) & ~(size_t)255;
    void* p = (char*)d_ws + off;
    off += bytes;
    return p;
  };
  float*     atoms  = (float*)alloc((size_t)5 * BL * 3 * 4);
  float*     Dnb    = (float*)alloc((size_t)NE * 4);
  int*       Eidx   = (int*)alloc((size_t)NE * 4);
  _Float16*  hE16   = (_Float16*)alloc((size_t)NE * HD * 2);
  _Float16*  msg16  = (_Float16*)alloc((size_t)NE * HD * 2);
  float*     hV32   = (float*)alloc((size_t)BL * HD * 4);
  _Float16*  hV16   = (_Float16*)alloc((size_t)BL * HD * 2);
  float*     hM32   = (float*)alloc((size_t)BL * HD * 4);
  _Float16*  hM16   = (_Float16*)alloc((size_t)BL * HD * 2);
  _Float16*  hVenc  = (_Float16*)alloc((size_t)BL * HD * 2);
  _Float16*  hS16   = (_Float16*)alloc((size_t)BL * HD * 2);
  int*       rank   = (int*)alloc((size_t)BL * 4);
  float*     mar    = (float*)alloc((size_t)NE * 4);
  auto aw = [&](size_t n) -> _Float16* { return (_Float16*)alloc(n * 2); };
  _Float16* edge_w16 = aw(416 * 128);
  _Float16* We_w16   = aw(128 * 128);
  struct EncW { _Float16 *w1,*w2,*w3,*w11,*w12,*w13,*fi,*fo; } ew[3];
  for (int l = 0; l < 3; ++l) {
    ew[l].w1 = aw(384 * 128); ew[l].w2 = aw(128 * 128); ew[l].w3 = aw(128 * 128);
    ew[l].w11 = aw(384 * 128); ew[l].w12 = aw(128 * 128); ew[l].w13 = aw(128 * 128);
    ew[l].fi = aw(128 * 512); ew[l].fo = aw(512 * 128);
  }
  struct DecW { _Float16 *w1,*w2,*w3,*fi,*fo; } dw[3];
  for (int l = 0; l < 3; ++l) {
    dw[l].w1 = aw(512 * 128); dw[l].w2 = aw(128 * 128); dw[l].w3 = aw(128 * 128);
    dw[l].fi = aw(128 * 512); dw[l].fo = aw(512 * 128);
  }

  auto cvt = [&](const float* s, _Float16* d, int din, int dout) {
    int n = din * dout;
    k_cvt_pack<<<(n + 255) / 256, 256, 0, stream>>>(s, d, din, dout);
  };
  cvt(edge_w, edge_w16, 416, 128);
  cvt(We_w, We_w16, 128, 128);
  for (int l = 0; l < 3; ++l) {
    cvt(ep[l].w1w, ew[l].w1, 384, 128);   cvt(ep[l].w2w, ew[l].w2, 128, 128);
    cvt(ep[l].w3w, ew[l].w3, 128, 128);   cvt(ep[l].w11w, ew[l].w11, 384, 128);
    cvt(ep[l].w12w, ew[l].w12, 128, 128); cvt(ep[l].w13w, ew[l].w13, 128, 128);
    cvt(ep[l].fiw, ew[l].fi, 128, 512);   cvt(ep[l].fow, ew[l].fo, 512, 128);
    cvt(dp[l].w1w, dw[l].w1, 512, 128);   cvt(dp[l].w2w, dw[l].w2, 128, 128);
    cvt(dp[l].w3w, dw[l].w3, 128, 128);   cvt(dp[l].fiw, dw[l].fi, 128, 512);
    cvt(dp[l].fow, dw[l].fo, 512, 128);
  }

  (void)hipMemsetAsync(hV32, 0, (size_t)BL * HD * 4, stream);
  (void)hipMemsetAsync(hV16, 0, (size_t)BL * HD * 2, stream);

  k_atoms<<<(BL + 255) / 256, 256, 0, stream>>>(X, atoms, BL);
  k_topk<<<BL, 256, 0, stream>>>(atoms, mask, Dnb, Eidx, L, BL);
  k_edge_features<<<NE / 16, 256, 0, stream>>>(atoms, Dnb, Eidx, residx, chains,
                                               pe_w, pe_b, edge_w16, ne_g, ne_b,
                                               We_w16, We_b, hE16, L, BL);
  for (int l = 0; l < 3; ++l) {
    k_enc_msg<<<NE / 16, 256, 0, stream>>>(hV16, hE16, Eidx, mask,
                                           ew[l].w1, ep[l].w1b, ew[l].w2, ep[l].w2b,
                                           ew[l].w3, ep[l].w3b, msg16, L);
    k_node_update<<<BL, 128, 0, stream>>>(msg16, hV32, ep[l].n1g, ep[l].n1b,
                                          hM32, hM16);
    k_ffn<<<BL / 16, 256, 0, stream>>>(hM16, hM32, ew[l].fi, ep[l].fib,
                                       ew[l].fo, ep[l].fob, ep[l].n2g, ep[l].n2b,
                                       mask, hV32, hV16);
    k_enc_edge<<<NE / 16, 256, 0, stream>>>(hV16, hE16, Eidx,
                                            ew[l].w11, ep[l].w11b, ew[l].w12,
                                            ep[l].w12b, ew[l].w13, ep[l].w13b,
                                            ep[l].n3g, ep[l].n3b, L);
  }
  k_hs<<<(BL * HD + 255) / 256, 256, 0, stream>>>(S, Ws, hS16, BL);
  k_rank<<<BL, 256, 0, stream>>>(chainM, mask, randn, rank, L);
  k_mar<<<(NE + 255) / 256, 256, 0, stream>>>(rank, Eidx, mar, L, NE);
  (void)hipMemcpyAsync(hVenc, hV16, (size_t)BL * HD * 2,
                       hipMemcpyDeviceToDevice, stream);
  for (int l = 0; l < 3; ++l) {
    k_dec_msg<<<NE / 16, 256, 0, stream>>>(hV16, hVenc, hE16, hS16, Eidx, mask, mar,
                                           dw[l].w1, dp[l].w1b, dw[l].w2, dp[l].w2b,
                                           dw[l].w3, dp[l].w3b, msg16, L);
    k_node_update<<<BL, 128, 0, stream>>>(msg16, hV32, dp[l].n1g, dp[l].n1b,
                                          hM32, hM16);
    k_ffn<<<BL / 16, 256, 0, stream>>>(hM16, hM32, dw[l].fi, dp[l].fib,
                                       dw[l].fo, dp[l].fob, dp[l].n2g, dp[l].n2b,
                                       mask, hV32, hV16);
  }
  k_out<<<BL, 32, 0, stream>>>(hV32, Wout_w, Wout_b, (float*)d_out);
}
